// HighSparsityMoE_16862041604487
// MI455X (gfx1250) — matrix operand order repaired and checked by reading, not compiled, not run on hardware
//
#include <hip/hip_runtime.h>
#include <hip/hip_bf16.h>

#define HIDDEN 896
#define INTER  2560
#define NEXP   16
#define TOPK   4
#define TOKENS 2048

#define TILE_M 32
#define TILE_N 128
#define TILE_K 32
#define LDSTR  40              // LDS row stride in bf16 elems (80B: 16B-aligned, bank-spread)
#define CAP    8704            // 8192 assignments + 16*31 worst-case padding, rounded to 32
#define MAXTILES (CAP / TILE_M) // 272

typedef __attribute__((ext_vector_type(16))) __bf16 v16bf;
typedef __attribute__((ext_vector_type(8)))  float  v8f;

union FragU { v16bf v; uint4 q[2]; };

__device__ inline unsigned short f2bf(float x) {
  unsigned int u = __float_as_uint(x);
  u = (u + 0x7fffu + ((u >> 16) & 1u)) >> 16;   // round-to-nearest-even
  return (unsigned short)u;
}

// A fragment 16x32 bf16 (M = lane%16): half h -> K = h*8..h*8+7 and 16+h*8..16+h*8+7
__device__ inline v16bf load_frag_a(const unsigned short* row, int half) {
  FragU f;
  f.q[0] = *reinterpret_cast<const uint4*>(row + half * 8);
  f.q[1] = *reinterpret_cast<const uint4*>(row + 16 + half * 8);
  return f.v;
}
// B fragment 32x16 bf16 stored transposed [N][K] in LDS (N = lane%16):
// half h -> K = h*8..h*8+7 and 16+h*8..16+h*8+7
__device__ inline v16bf load_frag_b(const unsigned short* row, int half) {
  FragU f;
  f.q[0] = *reinterpret_cast<const uint4*>(row + half * 8);
  f.q[1] = *reinterpret_cast<const uint4*>(row + 16 + half * 8);
  return f.v;
}

__device__ inline v8f wmma_bf16(v16bf a, v16bf b, v8f c) {
  return __builtin_amdgcn_wmma_f32_16x16x32_bf16(false, a, false, b, (short)0, c, false, false);
}

// Async copy of 8 bytes/lane from global to LDS (CDNA5 GLOBAL_LOAD_ASYNC_TO_LDS_B64).
__device__ inline void async_copy_b64(unsigned lds_byte_off, const void* gaddr) {
  asm volatile("global_load_async_to_lds_b64 %0, %1, off"
               :: "v"(lds_byte_off), "v"((unsigned long long)(size_t)gaddr)
               : "memory");
}
__device__ inline void wait_async0() {
  asm volatile("s_wait_asynccnt 0x0" ::: "memory");
}

// ---------------- router: logits -> softmax -> top4 -> renorm ----------------
__global__ __launch_bounds__(128) void router_kernel(
    const float* __restrict__ x, const float* __restrict__ rw,
    float* __restrict__ probs, int* __restrict__ top_i, float* __restrict__ top_w) {
  int t = blockIdx.x * 4 + (threadIdx.x >> 5);
  int lane = threadIdx.x & 31;
  const float* xr = x + (size_t)t * HIDDEN;
  float acc[NEXP];
#pragma unroll
  for (int e = 0; e < NEXP; ++e) acc[e] = 0.f;
  for (int d = lane; d < HIDDEN; d += 32) {
    float xv = xr[d];
    const float* w = rw + d * NEXP;
#pragma unroll
    for (int e = 0; e < NEXP; ++e) acc[e] = fmaf(xv, w[e], acc[e]);
  }
#pragma unroll
  for (int off = 16; off > 0; off >>= 1) {
#pragma unroll
    for (int e = 0; e < NEXP; ++e) acc[e] += __shfl_xor(acc[e], off, 32);
  }
  float mx = acc[0];
#pragma unroll
  for (int e = 1; e < NEXP; ++e) mx = fmaxf(mx, acc[e]);
  float p[NEXP]; float s = 0.f;
#pragma unroll
  for (int e = 0; e < NEXP; ++e) { p[e] = expf(acc[e] - mx); s += p[e]; }
  float inv = 1.f / s;
#pragma unroll
  for (int e = 0; e < NEXP; ++e) p[e] *= inv;
  if (lane < NEXP) probs[t * NEXP + lane] = p[lane];
  // top-4, ties -> lowest index first (matches jax.lax.top_k)
  float tv[TOPK]; int ti[TOPK]; float wsum = 0.f;
  bool used[NEXP];
#pragma unroll
  for (int e = 0; e < NEXP; ++e) used[e] = false;
#pragma unroll
  for (int k = 0; k < TOPK; ++k) {
    float bv = -1.f; int bi = 0;
#pragma unroll
    for (int e = 0; e < NEXP; ++e)
      if (!used[e] && p[e] > bv) { bv = p[e]; bi = e; }
    used[bi] = true; tv[k] = bv; ti[k] = bi; wsum += bv;
  }
  float rn = 1.f / (wsum + 1e-8f);
  if (lane == 0) {
#pragma unroll
    for (int k = 0; k < TOPK; ++k) {
      top_i[t * TOPK + k] = ti[k];
      top_w[t * TOPK + k] = tv[k] * rn;
    }
  }
}

// ---------------- fp32 -> bf16 activation convert ----------------
__global__ __launch_bounds__(256) void cvt_kernel(const float* __restrict__ x,
                                                  unsigned short* __restrict__ xb, int n) {
  int i = blockIdx.x * 256 + threadIdx.x;
  if (i < n) xb[i] = f2bf(x[i]);
}

// ---------------- counts, balance loss, segment offsets, tile tables ----------------
__global__ __launch_bounds__(256) void setup_kernel(
    const float* __restrict__ probs, const int* __restrict__ top_i,
    int* off, int* cursor, int* ntiles, int* tile_e, int* tile_b,
    int* assign_tk, float* loss_out) {
  __shared__ int cnt[NEXP];
  __shared__ float ps[NEXP];
  int tid = threadIdx.x;
  if (tid < NEXP) { cnt[tid] = 0; ps[tid] = 0.f; }
  __syncthreads();
  for (int i = tid; i < TOKENS * TOPK; i += 256) atomicAdd(&cnt[top_i[i]], 1);
  for (int i = tid; i < TOKENS * NEXP; i += 256) atomicAdd(&ps[i & (NEXP - 1)], probs[i]);
  for (int i = tid; i < CAP; i += 256) assign_tk[i] = -1;
  __syncthreads();
  if (tid == 0) {
    int o = 0, nt = 0;
    float L = 0.f;
    for (int e = 0; e < NEXP; ++e) {
      off[e] = o; cursor[e] = 0;
      int c = cnt[e];
      int pad = (c + TILE_M - 1) & ~(TILE_M - 1);
      for (int j = 0; j < pad / TILE_M; ++j) { tile_e[nt] = e; tile_b[nt] = o + j * TILE_M; ++nt; }
      o += pad;
      L += ((float)c / (float)TOKENS) * (ps[e] / (float)TOKENS);
    }
    off[NEXP] = o;
    *ntiles = nt;
    *loss_out = (float)NEXP * L;
  }
}

// ---------------- group assignments by expert ----------------
__global__ __launch_bounds__(256) void scatter_kernel(
    const int* __restrict__ top_i, const float* __restrict__ top_w,
    const int* __restrict__ off, int* cursor, int* assign_tk, float* assign_w) {
  int i = blockIdx.x * 256 + threadIdx.x;   // i = t*4 + k
  if (i >= TOKENS * TOPK) return;
  int e = top_i[i];
  int pos = atomicAdd(&cursor[e], 1);
  int a = off[e] + pos;
  assign_tk[a] = i;
  assign_w[a] = top_w[i];
}

// ---------------- GEMM1: h = silu(X*Wg) * (X*Wu), bf16 WMMA ----------------
// A tile staged with async global->LDS b64 copies; B tiles converted fp32->bf16
// via b128 loads + packed-pair ds_store_b32 into a transposed [n][k] layout.
// Padding rows read token 0 (garbage rows are discarded downstream).
__global__ __launch_bounds__(256) void gemm1_kernel(
    const unsigned short* __restrict__ xb,   // [TOKENS, HIDDEN] bf16
    const float* __restrict__ Wg, const float* __restrict__ Wu,   // [(E,)HIDDEN,INTER]
    unsigned short* __restrict__ hbuf,       // [rows, INTER] bf16
    const int* __restrict__ assign_tk,
    const int* __restrict__ tile_e, const int* __restrict__ tile_b,
    const int* __restrict__ ntiles, int mode) {
  int mt = blockIdx.x;
  int e = 0, base;
  if (mode) {
    if (mt >= *ntiles) return;
    e = tile_e[mt]; base = tile_b[mt];
  } else {
    base = mt * TILE_M;
  }
  const float* Bg = Wg + (size_t)e * HIDDEN * INTER;
  const float* Bu = Wu + (size_t)e * HIDDEN * INTER;
  int nbase = blockIdx.y * TILE_N;

  __shared__ __align__(16) unsigned short lA[TILE_M * LDSTR];
  __shared__ __align__(16) unsigned short lBg[TILE_N * LDSTR];
  __shared__ __align__(16) unsigned short lBu[TILE_N * LDSTR];

  int tid = threadIdx.x;
  // A staging plan: thread -> (row r, 8-byte chunk part); 256 threads cover 32x64B
  int ar = tid >> 3, ap = tid & 7;
  int tok;
  if (mode) {
    int tk = assign_tk[base + ar];
    tok = (tk >= 0) ? (tk >> 2) : 0;          // clamp padding to row 0 (discarded later)
  } else {
    tok = base + ar;
  }
  const unsigned short* agp = xb + (size_t)tok * HIDDEN + ap * 4;
  unsigned ldsA = (unsigned)(size_t)lA + (unsigned)(ar * (LDSTR * 2) + ap * 8);

  int wave = tid >> 5, lane = tid & 31;
  int msub = wave >> 2, nsub = wave & 3;      // 2 (M) x 4 (N) waves
  int c = lane & 15, half = lane >> 4;

  v8f cg0 = {}, cg1 = {}, cu0 = {}, cu1 = {};

  for (int k0 = 0; k0 < HIDDEN; k0 += TILE_K) {
    __syncthreads();
    async_copy_b64(ldsA, agp);                // A tile: 8 waves x 256B async
    agp += TILE_K;
    // B tiles: two adjacent K rows per thread, pack (k,k+1) bf16 pairs
    for (int i = tid; i < (TILE_K / 2) * (TILE_N / 4); i += 256) {  // 512 slots
      int kp = i >> 5;                // 0..15 -> k = 2*kp
      int n4 = (i & 31) << 2;         // 0,4,...,124
      size_t g0 = (size_t)(k0 + 2 * kp) * INTER + nbase + n4;
      float4 ga0 = *reinterpret_cast<const float4*>(Bg + g0);
      float4 ga1 = *reinterpret_cast<const float4*>(Bg + g0 + INTER);
      float4 gb0 = *reinterpret_cast<const float4*>(Bu + g0);
      float4 gb1 = *reinterpret_cast<const float4*>(Bu + g0 + INTER);
      int lo = 2 * kp;
      *reinterpret_cast<unsigned*>(&lBg[(n4 + 0) * LDSTR + lo]) = f2bf(ga0.x) | ((unsigned)f2bf(ga1.x) << 16);
      *reinterpret_cast<unsigned*>(&lBg[(n4 + 1) * LDSTR + lo]) = f2bf(ga0.y) | ((unsigned)f2bf(ga1.y) << 16);
      *reinterpret_cast<unsigned*>(&lBg[(n4 + 2) * LDSTR + lo]) = f2bf(ga0.z) | ((unsigned)f2bf(ga1.z) << 16);
      *reinterpret_cast<unsigned*>(&lBg[(n4 + 3) * LDSTR + lo]) = f2bf(ga0.w) | ((unsigned)f2bf(ga1.w) << 16);
      *reinterpret_cast<unsigned*>(&lBu[(n4 + 0) * LDSTR + lo]) = f2bf(gb0.x) | ((unsigned)f2bf(gb1.x) << 16);
      *reinterpret_cast<unsigned*>(&lBu[(n4 + 1) * LDSTR + lo]) = f2bf(gb0.y) | ((unsigned)f2bf(gb1.y) << 16);
      *reinterpret_cast<unsigned*>(&lBu[(n4 + 2) * LDSTR + lo]) = f2bf(gb0.z) | ((unsigned)f2bf(gb1.z) << 16);
      *reinterpret_cast<unsigned*>(&lBu[(n4 + 3) * LDSTR + lo]) = f2bf(gb0.w) | ((unsigned)f2bf(gb1.w) << 16);
    }
    wait_async0();                             // A-tile async done (this wave's writes)
    __syncthreads();
    v16bf a   = load_frag_a(&lA[(msub * 16 + c) * LDSTR], half);
    v16bf bg0 = load_frag_b(&lBg[(nsub * 32 + c) * LDSTR], half);
    v16bf bg1 = load_frag_b(&lBg[(nsub * 32 + 16 + c) * LDSTR], half);
    v16bf bu0 = load_frag_b(&lBu[(nsub * 32 + c) * LDSTR], half);
    v16bf bu1 = load_frag_b(&lBu[(nsub * 32 + 16 + c) * LDSTR], half);
    cg0 = wmma_bf16(a, bg0, cg0);
    cg1 = wmma_bf16(a, bg1, cg1);
    cu0 = wmma_bf16(a, bu0, cu0);
    cu1 = wmma_bf16(a, bu1, cu1);
  }

  int mrow0 = msub * 16 + half * 8;
  int ncol0 = nbase + nsub * 32 + c;
#pragma unroll
  for (int r = 0; r < 8; ++r) {
    int row = base + mrow0 + r;
    size_t o = (size_t)row * INTER + ncol0;
    float g0 = cg0[r], u0 = cu0[r];
    float g1 = cg1[r], u1 = cu1[r];
    hbuf[o]      = f2bf(g0 * u0 / (1.f + expf(-g0)));
    hbuf[o + 16] = f2bf(g1 * u1 / (1.f + expf(-g1)));
  }
}

// ---------------- GEMM2: y = (h * Wd) * route_weight ----------------
__global__ __launch_bounds__(256) void gemm2_kernel(
    const unsigned short* __restrict__ hbuf,  // [rows, INTER] bf16
    const float* __restrict__ Wd,             // [(E,)INTER, HIDDEN]
    float* __restrict__ outbuf,               // mode0: sout[T,D]; mode1: y[T*4,D]
    const int* __restrict__ assign_tk, const float* __restrict__ assign_w,
    const int* __restrict__ tile_e, const int* __restrict__ tile_b,
    const int* __restrict__ ntiles, int mode) {
  int mt = blockIdx.x;
  int e = 0, base;
  if (mode) {
    if (mt >= *ntiles) return;
    e = tile_e[mt]; base = tile_b[mt];
  } else {
    base = mt * TILE_M;
  }
  const float* B = Wd + (size_t)e * INTER * HIDDEN;
  int nbase = blockIdx.y * TILE_N;

  __shared__ __align__(16) unsigned short lA[TILE_M * LDSTR];
  __shared__ __align__(16) unsigned short lB[TILE_N * LDSTR];

  int tid = threadIdx.x;
  int ar = tid >> 3, ap = tid & 7;
  const unsigned short* agp = hbuf + (size_t)(base + ar) * INTER + ap * 4;
  unsigned ldsA = (unsigned)(size_t)lA + (unsigned)(ar * (LDSTR * 2) + ap * 8);

  int wave = tid >> 5, lane = tid & 31;
  int msub = wave >> 2, nsub = wave & 3;
  int c = lane & 15, half = lane >> 4;

  v8f c0 = {}, c1 = {};
  for (int k0 = 0; k0 < INTER; k0 += TILE_K) {
    __syncthreads();
    async_copy_b64(ldsA, agp);
    agp += TILE_K;
    for (int i = tid; i < (TILE_K / 2) * (TILE_N / 4); i += 256) {
      int kp = i >> 5;
      int n4 = (i & 31) << 2;
      size_t g0 = (size_t)(k0 + 2 * kp) * HIDDEN + nbase + n4;
      float4 b0 = *reinterpret_cast<const float4*>(B + g0);
      float4 b1 = *reinterpret_cast<const float4*>(B + g0 + HIDDEN);
      int lo = 2 * kp;
      *reinterpret_cast<unsigned*>(&lB[(n4 + 0) * LDSTR + lo]) = f2bf(b0.x) | ((unsigned)f2bf(b1.x) << 16);
      *reinterpret_cast<unsigned*>(&lB[(n4 + 1) * LDSTR + lo]) = f2bf(b0.y) | ((unsigned)f2bf(b1.y) << 16);
      *reinterpret_cast<unsigned*>(&lB[(n4 + 2) * LDSTR + lo]) = f2bf(b0.z) | ((unsigned)f2bf(b1.z) << 16);
      *reinterpret_cast<unsigned*>(&lB[(n4 + 3) * LDSTR + lo]) = f2bf(b0.w) | ((unsigned)f2bf(b1.w) << 16);
    }
    wait_async0();
    __syncthreads();
    v16bf a  = load_frag_a(&lA[(msub * 16 + c) * LDSTR], half);
    v16bf b0 = load_frag_b(&lB[(nsub * 32 + c) * LDSTR], half);
    v16bf b1 = load_frag_b(&lB[(nsub * 32 + 16 + c) * LDSTR], half);
    c0 = wmma_bf16(a, b0, c0);
    c1 = wmma_bf16(a, b1, c1);
  }

  int mrow0 = msub * 16 + half * 8;
  int ncol0 = nbase + nsub * 32 + c;
#pragma unroll
  for (int r = 0; r < 8; ++r) {
    int a_idx = base + mrow0 + r;
    if (mode) {
      int tk = assign_tk[a_idx];
      if (tk >= 0) {
        float w = assign_w[a_idx];
        outbuf[(size_t)tk * HIDDEN + ncol0]      = c0[r] * w;   // tk = t*4+k -> slot buffer
        outbuf[(size_t)tk * HIDDEN + ncol0 + 16] = c1[r] * w;
      }
    } else {
      outbuf[(size_t)a_idx * HIDDEN + ncol0]      = c0[r];
      outbuf[(size_t)a_idx * HIDDEN + ncol0 + 16] = c1[r];
    }
  }
}

// ---------------- combine: out = shared + sum of 4 routed slots ----------------
__global__ __launch_bounds__(256) void combine_kernel(
    const float* __restrict__ sout, const float* __restrict__ y, float* __restrict__ out) {
  int i = blockIdx.x * 256 + threadIdx.x;
  if (i >= TOKENS * HIDDEN) return;
  int t = i / HIDDEN, d = i - t * HIDDEN;
  float s = sout[i];
  size_t b = (size_t)t * TOPK * HIDDEN + d;
#pragma unroll
  for (int k = 0; k < TOPK; ++k) s += y[b + (size_t)k * HIDDEN];
  out[i] = s;
}

// ---------------- workspace layout ----------------
static constexpr size_t alup(size_t v) { return (v + 255) & ~(size_t)255; }
static constexpr size_t PROBS_OFF = 0;
static constexpr size_t TOPI_OFF  = alup(PROBS_OFF + sizeof(float) * TOKENS * NEXP);
static constexpr size_t TOPW_OFF  = alup(TOPI_OFF + sizeof(int) * TOKENS * TOPK);
static constexpr size_t OFF_OFF   = alup(TOPW_OFF + sizeof(float) * TOKENS * TOPK);
static constexpr size_t CUR_OFF   = alup(OFF_OFF + sizeof(int) * (NEXP + 1));
static constexpr size_t NT_OFF    = alup(CUR_OFF + sizeof(int) * NEXP);
static constexpr size_t TE_OFF    = alup(NT_OFF + sizeof(int));
static constexpr size_t TB_OFF    = alup(TE_OFF + sizeof(int) * MAXTILES);
static constexpr size_t ATK_OFF   = alup(TB_OFF + sizeof(int) * MAXTILES);
static constexpr size_t AW_OFF    = alup(ATK_OFF + sizeof(int) * CAP);
static constexpr size_t XB_OFF    = alup(AW_OFF + sizeof(float) * CAP);
static constexpr size_t HR_OFF    = alup(XB_OFF + sizeof(unsigned short) * TOKENS * HIDDEN);
static constexpr size_t HS_OFF    = alup(HR_OFF + sizeof(unsigned short) * (size_t)CAP * INTER);
static constexpr size_t Y_OFF     = alup(HS_OFF + sizeof(unsigned short) * (size_t)TOKENS * INTER);
static constexpr size_t SOUT_OFF  = alup(Y_OFF + sizeof(float) * (size_t)TOKENS * TOPK * HIDDEN);

extern "C" void kernel_launch(void* const* d_in, const int* in_sizes, int n_in,
                              void* d_out, int out_size, void* d_ws, size_t ws_size,
                              hipStream_t stream) {
  (void)in_sizes; (void)n_in; (void)out_size; (void)ws_size;
  const float* x  = (const float*)d_in[0];
  const float* rw = (const float*)d_in[1];
  const float* Wg = (const float*)d_in[2];
  const float* Wu = (const float*)d_in[3];
  const float* Wd = (const float*)d_in[4];
  const float* Sg = (const float*)d_in[5];
  const float* Su = (const float*)d_in[6];
  const float* Sd = (const float*)d_in[7];
  float* out = (float*)d_out;

  char* ws = (char*)d_ws;
  float*          probs  = (float*)(ws + PROBS_OFF);
  int*            top_i  = (int*)(ws + TOPI_OFF);
  float*          top_w  = (float*)(ws + TOPW_OFF);
  int*            off    = (int*)(ws + OFF_OFF);
  int*            cursor = (int*)(ws + CUR_OFF);
  int*            ntiles = (int*)(ws + NT_OFF);
  int*            tile_e = (int*)(ws + TE_OFF);
  int*            tile_b = (int*)(ws + TB_OFF);
  int*            atk    = (int*)(ws + ATK_OFF);
  float*          aw     = (float*)(ws + AW_OFF);
  unsigned short* xb     = (unsigned short*)(ws + XB_OFF);
  unsigned short* hr     = (unsigned short*)(ws + HR_OFF);
  unsigned short* hs     = (unsigned short*)(ws + HS_OFF);
  float*          y      = (float*)(ws + Y_OFF);
  float*          sout   = (float*)(ws + SOUT_OFF);

  router_kernel<<<TOKENS / 4, 128, 0, stream>>>(x, rw, probs, top_i, top_w);
  cvt_kernel<<<(TOKENS * HIDDEN + 255) / 256, 256, 0, stream>>>(x, xb, TOKENS * HIDDEN);
  setup_kernel<<<1, 256, 0, stream>>>(probs, top_i, off, cursor, ntiles, tile_e, tile_b,
                                      atk, out + (size_t)TOKENS * HIDDEN);
  scatter_kernel<<<(TOKENS * TOPK + 255) / 256, 256, 0, stream>>>(top_i, top_w, off, cursor, atk, aw);

  dim3 g1s(TOKENS / TILE_M, INTER / TILE_N);   // 64 x 20
  gemm1_kernel<<<g1s, 256, 0, stream>>>(xb, Sg, Su, hs, nullptr, nullptr, nullptr, nullptr, 0);
  dim3 g1r(MAXTILES, INTER / TILE_N);          // 272 x 20
  gemm1_kernel<<<g1r, 256, 0, stream>>>(xb, Wg, Wu, hr, atk, tile_e, tile_b, ntiles, 1);

  dim3 g2s(TOKENS / TILE_M, HIDDEN / TILE_N);  // 64 x 7
  gemm2_kernel<<<g2s, 256, 0, stream>>>(hs, Sd, sout, nullptr, nullptr, nullptr, nullptr, nullptr, 0);
  dim3 g2r(MAXTILES, HIDDEN / TILE_N);         // 272 x 7
  gemm2_kernel<<<g2r, 256, 0, stream>>>(hr, Wd, y, atk, aw, tile_e, tile_b, ntiles, 1);

  combine_kernel<<<(TOKENS * HIDDEN + 255) / 256, 256, 0, stream>>>(sout, y, out);
}
